// VFSE23vbiasBase_82076825027091
// MI455X (gfx1250) — matrix operand order repaired and checked by reading, not compiled, not run on hardware
//
#include <hip/hip_runtime.h>
#include <hip/hip_bf16.h>

typedef __attribute__((ext_vector_type(16))) _Float16 v16h;
typedef __attribute__((ext_vector_type(8)))  _Float16 v8h;
typedef __attribute__((ext_vector_type(8)))  float    v8f;

#define MT 4                 // M-tiles (of 16 rows) per block -> 64 rows/block
#define ABUF_W 776           // 768 (=256+512 max in+out widths) + 8 pad halves
                             // row stride 1552 B: 16B-aligned, rows on distinct bank groups

struct FusedArgs {
  const float* y;
  const float* imu;
  const float* imud;
  const float* R0;
  const _Float16* w[13];  // transposed, zero-padded f16 weights [Npad][Kpad]
  const float* b[13];     // f32 biases straight from d_in
  float* out;
};

__device__ __forceinline__ float act_gelu(float x) {
  return 0.5f * x * (1.0f + erff(x * 0.70710678118654752f));
}

// One GEMM layer over MT 16-row tiles; 8 waves split the (m,n) tile space,
// m interleaved across waves so narrow layers (1 n-tile) still use MT waves.
// A layout (16x32 f16): lane<16 -> M=lane, K k..k+15; lane>=16 -> M=lane-16, K k+16..k+31.
// B layout (32x16 f16): lane<16 -> col=lane, K k..k+15; lane>=16 -> col=lane-16, K k+16..k+31.
// D layout: col = lane&15, row = (lane>>4)*8 + r.
template<int KPAD, int ACT>   // ACT: 0=linear, 1=gelu, 2=tanh
__device__ __forceinline__ void layer(
    const _Float16* __restrict__ in, int sin,
    _Float16* __restrict__ outb, int sout,
    const _Float16* __restrict__ Wt,
    const float* __restrict__ bias, int N,
    const _Float16* xadd,   // residual (LDS, stride 32) or nullptr
    float* gout)            // global output base (row stride 16, first 3 cols) or nullptr
{
  const int lane = threadIdx.x & 31;
  const int wave = threadIdx.x >> 5;
  const int col  = lane & 15;
  const int hi   = lane >> 4;
  const int n_tiles = (N + 15) >> 4;
  const int total = n_tiles * MT;

  for (int t = wave; t < total; t += 8) {
    const int mt = t & (MT - 1);
    const int nt = t >> 2;               // log2(MT)
    const int row0 = mt * 16;
    const _Float16* arow = in + (size_t)(row0 + col) * sin + hi * 16;
    const _Float16* wcol = Wt + (size_t)(nt * 16 + col) * KPAD + hi * 16;
    v8f acc = {};
#pragma unroll
    for (int k = 0; k < KPAD; k += 32) {
      union { v16h v; v8h h[2]; } a;
      a.h[0] = *(const v8h*)(arow + k);          // ds_load_b128
      a.h[1] = *(const v8h*)(arow + k + 8);      // ds_load_b128
      v16h bfrag = *(const v16h*)(wcol + k);     // 32B global load (column-major weights)
      acc = __builtin_amdgcn_wmma_f32_16x16x32_f16(false, a.v, false, bfrag,
                                                   (short)0, acc, false, false);
    }
    const int n = nt * 16 + col;
    const float bv = (n < N) ? bias[n] : 0.0f;
#pragma unroll
    for (int r = 0; r < 8; ++r) {
      const int row = row0 + hi * 8 + r;
      float v = acc[r] + bv;
      if (ACT == 1) v = act_gelu(v);
      if (ACT == 2) v = tanhf(v);
      if (xadd) v += (float)xadd[row * 32 + col];
      outb[row * sout + nt * 16 + col] = (_Float16)v;
      if (gout && col < 3) gout[row * 16 + col] = v;
    }
  }
  __syncthreads();
}

__global__ void prep_w(const float* __restrict__ src, _Float16* __restrict__ dst,
                       int K, int N, int Kpad, int Npad) {
  int idx = blockIdx.x * blockDim.x + threadIdx.x;
  if (idx >= Kpad * Npad) return;
  int k = idx % Kpad, n = idx / Kpad;
  float v = (k < K && n < N) ? src[k * N + n] : 0.0f;  // src is (K,N) row-major
  dst[idx] = (_Float16)v;
}

__global__ __launch_bounds__(256) void fused_kernel(FusedArgs A) {
  __shared__ _Float16 ABUF[MT * 16 * ABUF_W];  // 99,328 B activation workspace
  __shared__ _Float16 sX[MT * 16 * 32];        // 9-wide residual stream, K-padded to 32

  const int base = blockIdx.x * (MT * 16);
  float* outB = A.out + (size_t)base * 16;

  // ---------------- per-row SO(3) geometry: threads 0..63 (waves 0-1, full EXEC) --------
  if (threadIdx.x < MT * 16) {
    const int g = base + (int)threadIdx.x;
    const float* yr = A.y + (size_t)g * 16;
    const float* ir = A.imu + (size_t)g * 6;
    float px = yr[0], py = yr[1], pz = yr[2];
    float th2 = px * px + py * py + pz * pz;
    bool small = th2 < 1e-8f;
    float th2s = small ? 1.0f : th2;
    float th = sqrtf(th2s);
    float cA, cB, cJ;
    if (small) {
      cA = 1.0f - th2 / 6.0f;
      cB = 0.5f - th2 / 24.0f;
      cJ = 1.0f / 12.0f + th2 / 720.0f;
    } else {
      float s = sinf(th), c = cosf(th);
      cA = s / th;
      cB = (1.0f - c) / th2s;
      cJ = 1.0f / th2s - (1.0f + c) / (2.0f * th * s);
    }
    float ph[3] = {px, py, pz};
    float K[3][3] = {{0.f, -pz, py}, {pz, 0.f, -px}, {-py, px, 0.f}};
    float R[3][3], Jinv[3][3];
#pragma unroll
    for (int i = 0; i < 3; ++i)
#pragma unroll
      for (int j = 0; j < 3; ++j) {
        float k2 = ph[i] * ph[j] - (i == j ? th2 : 0.0f);   // (K@K)[i][j]
        float id = (i == j) ? 1.0f : 0.0f;
        R[i][j]    = id + cA * K[i][j] + cB * k2;
        Jinv[i][j] = id + 0.5f * K[i][j] + cJ * k2;
      }
    float wm[3] = {ir[0] - yr[4],  ir[1] - yr[5],  ir[2] - yr[6]};
    float am[3] = {ir[3] - yr[13], ir[4] - yr[14], ir[5] - yr[15]};
    float* o = outB + threadIdx.x * 16;
#pragma unroll
    for (int i = 0; i < 3; ++i)
      o[i] = Jinv[i][0] * wm[0] + Jinv[i][1] * wm[1] + Jinv[i][2] * wm[2];
    o[3] = 1.0f;
    float Ra[3];
#pragma unroll
    for (int k = 0; k < 3; ++k)
      Ra[k] = R[k][0] * am[0] + R[k][1] * am[1] + R[k][2] * am[2];
#pragma unroll
    for (int i = 0; i < 3; ++i) {
      float v = A.R0[i * 3 + 0] * Ra[0] + A.R0[i * 3 + 1] * Ra[1] + A.R0[i * 3 + 2] * Ra[2];
      if (i == 2) v -= 9.81f;
      o[7 + i]  = v;
      o[10 + i] = yr[7 + i];
    }
  }

  // ---------------- bw_net ----------------
  for (int idx = threadIdx.x; idx < MT * 16 * 32; idx += 256) {
    int row = idx >> 5, c = idx & 31;
    const int g = base + row;
    float v = 0.0f;
    if (c < 3)      v = A.y[(size_t)g * 16 + 4 + c];
    else if (c < 6) v = A.imu[(size_t)g * 6 + (c - 3)];
    else if (c < 9) v = A.imud[(size_t)g * 6 + (c - 6)];
    sX[row * 32 + c] = (_Float16)v;
  }
  __syncthreads();

  layer<32,  1>(sX,         32,     ABUF,       ABUF_W, A.w[0], A.b[0], 128, nullptr, nullptr);
  layer<128, 2>(ABUF,       ABUF_W, ABUF + 128, ABUF_W, A.w[1], A.b[1], 128, nullptr, nullptr);
  layer<128, 0>(ABUF + 128, ABUF_W, sX,         32,     A.w[2], A.b[2], 9,   sX,      nullptr);
  layer<32,  2>(sX,         32,     ABUF,       ABUF_W, A.w[3], A.b[3], 512, nullptr, nullptr);
  layer<512, 0>(ABUF,       ABUF_W, sX,         32,     A.w[4], A.b[4], 9,   sX,      nullptr);
  layer<32,  0>(sX,         32,     ABUF,       ABUF_W, A.w[5], A.b[5], 3,   nullptr, outB + 4);

  // ---------------- ba_net ----------------
  for (int idx = threadIdx.x; idx < MT * 16 * 32; idx += 256) {
    int row = idx >> 5, c = idx & 31;
    const int g = base + row;
    float v = 0.0f;
    if (c < 3)      v = A.y[(size_t)g * 16 + 13 + c];
    else if (c < 6) v = A.imu[(size_t)g * 6 + 3 + (c - 3)];
    else if (c < 9) v = A.imud[(size_t)g * 6 + 3 + (c - 6)];
    sX[row * 32 + c] = (_Float16)v;
  }
  __syncthreads();

  layer<32,  1>(sX,         32,     ABUF,       ABUF_W, A.w[6],  A.b[6],  256, nullptr, nullptr);
  layer<256, 1>(ABUF,       ABUF_W, ABUF + 256, ABUF_W, A.w[7],  A.b[7],  512, nullptr, nullptr);
  layer<512, 0>(ABUF + 256, ABUF_W, sX,         32,     A.w[8],  A.b[8],  9,   sX,      nullptr);
  layer<32,  1>(sX,         32,     ABUF,       ABUF_W, A.w[9],  A.b[9],  256, nullptr, nullptr);
  layer<256, 1>(ABUF,       ABUF_W, ABUF + 256, ABUF_W, A.w[10], A.b[10], 256, nullptr, nullptr);
  layer<256, 0>(ABUF + 256, ABUF_W, sX,         32,     A.w[11], A.b[11], 9,   sX,      nullptr);
  layer<32,  0>(sX,         32,     ABUF,       ABUF_W, A.w[12], A.b[12], 3,   nullptr, outB + 13);
}

extern "C" void kernel_launch(void* const* d_in, const int* in_sizes, int n_in,
                              void* d_out, int out_size, void* d_ws, size_t ws_size,
                              hipStream_t stream) {
  (void)n_in; (void)out_size; (void)ws_size;
  // d_in layout: 0=y 1=imu 2=imud, 3..14 = bw_params (W,b interleaved),
  //              15..28 = ba_params, 29 = R0
  static const int KDIMS[13] = {9, 128, 128, 9, 512, 9,   9, 256, 512, 9, 256, 256, 9};
  static const int NDIMS[13] = {128, 128, 9, 512, 9, 3,   256, 512, 9, 256, 256, 9, 3};
  static const int WIDX[13]  = {3, 5, 7, 9, 11, 13,       15, 17, 19, 21, 23, 25, 27};

  _Float16* ws = (_Float16*)d_ws;
  FusedArgs A;
  A.y    = (const float*)d_in[0];
  A.imu  = (const float*)d_in[1];
  A.imud = (const float*)d_in[2];
  A.R0   = (const float*)d_in[29];
  A.out  = (float*)d_out;

  size_t off = 0;
  for (int i = 0; i < 13; ++i) {
    int K = KDIMS[i], N = NDIMS[i];
    int Kpad = (K + 31) & ~31;
    int Npad = (N + 15) & ~15;
    int total = Kpad * Npad;
    prep_w<<<(total + 255) / 256, 256, 0, stream>>>(
        (const float*)d_in[WIDX[i]], ws + off, K, N, Kpad, Npad);
    A.w[i] = ws + off;
    A.b[i] = (const float*)d_in[WIDX[i] + 1];
    off += (size_t)total;
  }

  const int rows = in_sizes[0] / 16;       // 131072
  fused_kernel<<<rows / (MT * 16), 256, 0, stream>>>(A);
}
